// CausePredictor_16638703305436
// MI455X (gfx1250) — matrix operand order repaired and checked by reading, not compiled, not run on hardware
//
#include <hip/hip_runtime.h>
#include <hip/hip_bf16.h>
#include <math.h>

typedef __attribute__((ext_vector_type(16))) _Float16 v16h;
typedef __attribute__((ext_vector_type(8)))  _Float16 v8h;
typedef __attribute__((ext_vector_type(8)))  float    v8f;

#define B_ 4
#define S_ 200
#define D_ 300
#define M_ 512
#define P_ 100
#define NREL_ 9
#define MAXLEN_ 10
#define LDP 520          // padded LDS row stride in halves (1040B => 4-bank skew, b128-aligned)
#define TJ 64            // j-rows per workgroup (4 x 16-row A sub-tiles)
#define NJT 4            // ceil(200/64)

__device__ __forceinline__ int rel_of(int d) {
    // d = i - j
    if (d == 0) return 0;
    if (d < 0) return 1;
    int q = (d + 1) >> 1;       // ceil(d/2)
    if (q > 8) q = 8;           // clamp at -(WINDOW+1) = -8
    return 9 - q;               // (-q) mod 9
}

// ---- kernel 1: inv_deg[r][j] -------------------------------------------------
__global__ void k_deg(float* invdeg) {
    int j = blockIdx.x * blockDim.x + threadIdx.x;
    if (j >= S_) return;
    int cnt[NREL_];
#pragma unroll
    for (int r = 0; r < NREL_; ++r) cnt[r] = 0;
    for (int i = 0; i < S_; ++i) cnt[rel_of(i - j)]++;
#pragma unroll
    for (int r = 0; r < NREL_; ++r)
        invdeg[r * S_ + j] = cnt[r] > 0 ? 1.0f / (float)cnt[r] : 0.0f;
}

// ---- kernel 2: per-(i,j) scalar coefficients --------------------------------
__global__ void k_coeff(const float* __restrict__ comp, const float* __restrict__ invdeg,
                        float* __restrict__ c0, float* __restrict__ c1) {
    int t = blockIdx.x * blockDim.x + threadIdx.x;
    if (t >= S_ * S_) return;
    int i = t / S_, j = t % S_;
    int r = rel_of(i - j);
    float idg = invdeg[r * S_ + j];
    c0[t] = comp[r * 2 + 0] * idg;
    c1[t] = comp[r * 2 + 1] * idg;
}

// ---- kernel 3: m0/m1[b,j,d] = sum_i c[i,j] * x[b,i,d] -----------------------
__global__ void k_msg(const float* __restrict__ x, const float* __restrict__ c0,
                      const float* __restrict__ c1, float* __restrict__ m0,
                      float* __restrict__ m1) {
    int t = blockIdx.x * blockDim.x + threadIdx.x;
    if (t >= B_ * S_ * D_) return;
    int d = t % D_;
    int n = t / D_;          // n = b*S + j
    int b = n / S_, j = n % S_;
    float a0 = 0.f, a1 = 0.f;
    const float* xb = x + (size_t)b * S_ * D_ + d;
    for (int i = 0; i < S_; ++i) {
        float xv = xb[(size_t)i * D_];
        a0 += c0[i * S_ + j] * xv;
        a1 += c1[i * S_ + j] * xv;
    }
    m0[t] = a0; m1[t] = a1;
}

// ---- kernel 4: h[b,j,e] = m0@basis0 + m1@basis1 + x@root + bias -------------
__global__ void k_h(const float* __restrict__ x, const float* __restrict__ m0,
                    const float* __restrict__ m1, const float* __restrict__ basis,
                    const float* __restrict__ root, const float* __restrict__ bias,
                    float* __restrict__ h) {
    int t = blockIdx.x * blockDim.x + threadIdx.x;
    if (t >= B_ * S_ * D_) return;
    int e = t % D_;
    int n = t / D_;
    const float* b0 = basis;            // (D,D)
    const float* b1 = basis + D_ * D_;
    float acc = bias[e];
    const float* mr0 = m0 + (size_t)n * D_;
    const float* mr1 = m1 + (size_t)n * D_;
    const float* xr  = x  + (size_t)n * D_;
    for (int d = 0; d < D_; ++d) {
        acc += mr0[d] * b0[d * D_ + e];
        acc += mr1[d] * b1[d * D_ + e];
        acc += xr[d]  * root[d * D_ + e];
    }
    h[t] = acc;
}

// ---- kernel 5: Ha = h@W1a, Hc = h@W1c ---------------------------------------
__global__ void k_hw1(const float* __restrict__ h, const float* __restrict__ W1,
                      float* __restrict__ Ha, float* __restrict__ Hc) {
    int t = blockIdx.x * blockDim.x + threadIdx.x;
    if (t >= B_ * S_ * M_) return;
    int m = t % M_;
    int n = t / M_;
    const float* hr = h + (size_t)n * D_;
    const float* wa = W1 + m;                          // rows [0,D)
    const float* wc = W1 + (size_t)(D_ + P_) * M_ + m; // rows [D+P, 2D+P)
    float aa = 0.f, ac = 0.f;
    for (int d = 0; d < D_; ++d) {
        float hv = hr[d];
        aa += hv * wa[(size_t)d * M_];
        ac += hv * wc[(size_t)d * M_];
    }
    Ha[t] = aa; Hc[t] = ac;
}

// ---- kernel 6: PeTab[p,m] = pe_k[p]@W1b + pe_v[p]@W1d -----------------------
__global__ void k_petab(const float* __restrict__ pe_k, const float* __restrict__ pe_v,
                        const float* __restrict__ W1, float* __restrict__ PeTab) {
    int t = blockIdx.x * blockDim.x + threadIdx.x;
    if (t >= (MAXLEN_ + 1) * M_) return;
    int m = t % M_;
    int p = t / M_;
    const float* wb = W1 + (size_t)D_ * M_ + m;             // rows [D, D+P)
    const float* wd = W1 + (size_t)(2 * D_ + P_) * M_ + m;  // rows [2D+P, 2D+2P)
    float acc = 0.f;
    for (int q = 0; q < P_; ++q)
        acc += pe_k[p * P_ + q] * wb[(size_t)q * M_] +
               pe_v[p * P_ + q] * wd[(size_t)q * M_];
    PeTab[t] = acc;
}

// ---- kernel 7: pack W2 (f32 MxM row-major) into f16 B-fragment layout -------
// layout: frag[((nt*16 + kt)*32 + lane)*16 + e] ; k = kt*32 + (lane>=16?16:0)+e ;
//         n = nt*16 + (lane&15)
__global__ void k_packW2(const float* __restrict__ W2, _Float16* __restrict__ W2h) {
    int t = blockIdx.x * blockDim.x + threadIdx.x;
    if (t >= M_ * M_) return;
    int e    = t & 15;
    int lane = (t >> 4) & 31;
    int kt   = (t >> 9) & 15;
    int nt   = t >> 13;
    int k = kt * 32 + ((lane >> 4) << 4) + e;
    int n = nt * 16 + (lane & 15);
    W2h[t] = (_Float16)W2[(size_t)k * M_ + n];
}

// ---- kernel 8: fused row-build + 64x512 @ 512x512 WMMA GEMM + relu + @Wp ----
// Each workgroup: one (b,i), 64 j-rows (4 A sub-tiles of 16).  8 waves own
// N-tiles {w, w+8, w+16, w+24}; each pair of B-fragment loads feeds 4 A-tiles
// (16 WMMAs per kt per wave) => 4x better B reuse than a 16-row tile.
__global__ void __launch_bounds__(256)
k_main(const float* __restrict__ Ha, const float* __restrict__ Hc,
       const float* __restrict__ PeTab, const _Float16* __restrict__ W2h,
       const float* __restrict__ Wp, const float* __restrict__ mask,
       float* __restrict__ out) {
    extern __shared__ char smem[];
    _Float16* Arows = (_Float16*)smem;                       // TJ * LDP halves
    float*    accs  = (float*)(smem + TJ * LDP * 2);         // TJ floats

    const int tile = blockIdx.x;   // j tile (TJ rows)
    const int i    = blockIdx.y;   // query index (axis "a")
    const int b    = blockIdx.z;
    const int j0   = tile * TJ;
    const int t    = threadIdx.x;

    if (t < TJ) accs[t] = 0.0f;

    // --- build TJ rows of relu(Ha[b,j] + Hc[b,i] + PeTab[pos]) in f16 LDS ---
    {
        const int r  = t >> 2;            // row 0..63
        const int kb = (t & 3) * 128;     // 128 columns per thread
        const int j  = j0 + r;
        if (j < S_) {
            int pos = i - j + 1;
            pos = pos < 0 ? 0 : (pos > MAXLEN_ ? MAXLEN_ : pos);
            const float* ha = Ha + (size_t)(b * S_ + j) * M_;
            const float* hc = Hc + (size_t)(b * S_ + i) * M_;
            const float* pe = PeTab + (size_t)pos * M_;
            for (int k = kb; k < kb + 128; ++k) {
                float v = ha[k] + hc[k] + pe[k];
                Arows[r * LDP + k] = (_Float16)(v > 0.f ? v : 0.f);
            }
        } else {
            for (int k = kb; k < kb + 128; ++k)
                Arows[r * LDP + k] = (_Float16)0.f;
        }
    }
    __syncthreads();

    const int wave = t >> 5;
    const int lane = t & 31;
    const int arow = lane & 15;
    const int akb  = (lane >> 4) * 16;    // A K-chunk base within 32-K tile

    v8f c[4][4];                          // [a-subtile][n-tile]
#pragma unroll
    for (int at = 0; at < 4; ++at)
#pragma unroll
        for (int q = 0; q < 4; ++q)
            c[at][q] = (v8f){};

    for (int kt = 0; kt < 16; ++kt) {
        const int ko = kt * 32 + akb;
        const _Float16* bbase = W2h + ((size_t)kt * 32 + lane) * 16;
        // one set of B fragments per kt, reused across 4 A sub-tiles
        v16h bf[4];
#pragma unroll
        for (int q = 0; q < 4; ++q)
            bf[q] = *(const v16h*)(bbase + (size_t)(wave + q * 8) * 8192);
#pragma unroll
        for (int at = 0; at < 4; ++at) {
            const int row = at * 16 + arow;
            v8h a0 = *(const v8h*)&Arows[row * LDP + ko];
            v8h a1 = *(const v8h*)&Arows[row * LDP + ko + 8];
            v16h a = __builtin_shufflevector(a0, a1, 0, 1, 2, 3, 4, 5, 6, 7,
                                                     8, 9, 10, 11, 12, 13, 14, 15);
#pragma unroll
            for (int q = 0; q < 4; ++q)
                c[at][q] = __builtin_amdgcn_wmma_f32_16x16x32_f16(
                    false, a, false, bf[q], (short)0, c[at][q], false, false);
        }
    }

    // --- fused relu + dot with Wp, reduce C-tile columns ---------------------
    {
        const int ncol = lane & 15;
        float w[4];
#pragma unroll
        for (int q = 0; q < 4; ++q)
            w[q] = Wp[(wave + q * 8) * 16 + ncol];

#pragma unroll
        for (int at = 0; at < 4; ++at) {
            float p[8];
#pragma unroll
            for (int v = 0; v < 8; ++v) {
                float acc = 0.f;
#pragma unroll
                for (int q = 0; q < 4; ++q) {
                    float e = c[at][q][v];
                    e = e > 0.f ? e : 0.f;
                    acc = fmaf(e, w[q], acc);
                }
                p[v] = acc;
            }
            // reduce across the 16 lanes of each half-wave (tile columns)
#pragma unroll
            for (int off = 1; off < 16; off <<= 1)
#pragma unroll
                for (int v = 0; v < 8; ++v)
                    p[v] += __shfl_xor(p[v], off, 32);

            if ((lane & 15) == 0) {
                const int rbase = at * 16 + (lane >> 4) * 8; // lanes<16: M=v, lanes>=16: M=v+8
#pragma unroll
                for (int v = 0; v < 8; ++v)
                    atomicAdd(&accs[rbase + v], p[v]);
            }
        }
    }
    __syncthreads();

    if (t < TJ) {
        const int j = j0 + t;
        if (j < S_) {
            const size_t oidx = ((size_t)b * S_ + i) * S_ + j;
            float u = accs[t];
            float s = 1.0f / (1.0f + __expf(-u));
            out[oidx] = s * mask[oidx];
        }
    }
}

extern "C" void kernel_launch(void* const* d_in, const int* in_sizes, int n_in,
                              void* d_out, int out_size, void* d_ws, size_t ws_size,
                              hipStream_t stream) {
    const float* x     = (const float*)d_in[0];
    const float* mask  = (const float*)d_in[1];
    const float* pe_k  = (const float*)d_in[2];
    const float* pe_v  = (const float*)d_in[3];
    const float* comp  = (const float*)d_in[4];
    const float* basis = (const float*)d_in[5];
    const float* root  = (const float*)d_in[6];
    const float* bias  = (const float*)d_in[7];
    const float* W1    = (const float*)d_in[8];
    const float* W2    = (const float*)d_in[9];
    const float* Wp    = (const float*)d_in[10];
    float* out = (float*)d_out;

    // workspace carve-up (64B aligned)
    char* ws = (char*)d_ws;
    size_t off = 0;
    auto alloc = [&](size_t bytes) {
        char* p = ws + off;
        off += (bytes + 63) & ~(size_t)63;
        return p;
    };
    float*    invdeg = (float*)alloc(NREL_ * S_ * sizeof(float));
    float*    c0     = (float*)alloc(S_ * S_ * sizeof(float));
    float*    c1     = (float*)alloc(S_ * S_ * sizeof(float));
    float*    m0     = (float*)alloc((size_t)B_ * S_ * D_ * sizeof(float));
    float*    m1     = (float*)alloc((size_t)B_ * S_ * D_ * sizeof(float));
    float*    h      = (float*)alloc((size_t)B_ * S_ * D_ * sizeof(float));
    float*    Ha     = (float*)alloc((size_t)B_ * S_ * M_ * sizeof(float));
    float*    Hc     = (float*)alloc((size_t)B_ * S_ * M_ * sizeof(float));
    float*    PeTab  = (float*)alloc((size_t)(MAXLEN_ + 1) * M_ * sizeof(float));
    _Float16* W2h    = (_Float16*)alloc((size_t)M_ * M_ * sizeof(_Float16));

    const int TB = 256;
    k_deg<<<1, TB, 0, stream>>>(invdeg);
    k_coeff<<<(S_ * S_ + TB - 1) / TB, TB, 0, stream>>>(comp, invdeg, c0, c1);
    k_msg<<<(B_ * S_ * D_ + TB - 1) / TB, TB, 0, stream>>>(x, c0, c1, m0, m1);
    k_h<<<(B_ * S_ * D_ + TB - 1) / TB, TB, 0, stream>>>(x, m0, m1, basis, root, bias, h);
    k_hw1<<<(B_ * S_ * M_ + TB - 1) / TB, TB, 0, stream>>>(h, W1, Ha, Hc);
    k_petab<<<((MAXLEN_ + 1) * M_ + TB - 1) / TB, TB, 0, stream>>>(pe_k, pe_v, W1, PeTab);
    k_packW2<<<(M_ * M_ + TB - 1) / TB, TB, 0, stream>>>(W2, W2h);

    const size_t lds_bytes = (size_t)TJ * LDP * sizeof(_Float16) + TJ * sizeof(float);
    k_main<<<dim3(NJT, S_, B_), TB, lds_bytes, stream>>>(Ha, Hc, PeTab, W2h, Wp, mask, out);
}
